// KGATNE_58196806861278
// MI455X (gfx1250) — compile-verified, matrix operand order checked by reading, not run on hardware
//
#include <hip/hip_runtime.h>

#define DD   100      // feature dim
#define KPAD 128      // padded K for all GEMMs (100 -> 128)

typedef __attribute__((ext_vector_type(16))) __bf16 v16bf;
typedef __attribute__((ext_vector_type(8)))  float  v8f;

__device__ __forceinline__ __bf16 f2bf(float x) { return (__bf16)x; }

// Pack 8 floats (two float4 runs) into fragment elements [base, base+8)
__device__ __forceinline__ void pack8(v16bf& f, int base, float4 a, float4 b) {
  f[base + 0] = f2bf(a.x); f[base + 1] = f2bf(a.y);
  f[base + 2] = f2bf(a.z); f[base + 3] = f2bf(a.w);
  f[base + 4] = f2bf(b.x); f[base + 5] = f2bf(b.y);
  f[base + 6] = f2bf(b.z); f[base + 7] = f2bf(b.w);
}

// Fragment element -> k-local mapping per documented 16-bit 16x32 layout.
__device__ __forceinline__ int frag_klocal(int ii, int hi) {
  int vv = ii >> 1, e = ii & 1;
  return (vv < 4) ? (2 * vv + e + 8 * hi) : (16 + 2 * (vv - 4) + e + 8 * hi);
}

// ---------------------------------------------------------------------------
// Weight convert f32->bf16, pad to [KPAD x ntiles*16], emitted PRE-SWIZZLED
// into WMMA B-fragment order: dst[((nt*4 + s)*32 + lane)*16 + ii].
// ---------------------------------------------------------------------------
__global__ void padswz_kernel(const float* __restrict__ src, int Ka, int Na,
                              __bf16* __restrict__ dst, int ntiles) {
  int i = blockIdx.x * blockDim.x + threadIdx.x;
  if (i >= ntiles * 2048) return;           // ntiles * 4 ksteps * 32 lanes * 16
  int ii = i & 15;
  int ln = (i >> 4) & 31;
  int s  = (i >> 9) & 3;
  int nt = i >> 11;
  int m = ln & 15, hi = ln >> 4;
  int k = 32 * s + frag_klocal(ii, hi);
  int n = nt * 16 + m;
  float v = (k < Ka && n < Na) ? src[(size_t)k * Na + n] : 0.0f;
  dst[i] = f2bf(v);
}

// ---------------------------------------------------------------------------
// out[r][:] = emb[idx[r]][:]   (float4-wide, 25 vec4 per 100-float row)
// ---------------------------------------------------------------------------
__global__ void gather_kernel(const float* __restrict__ emb,
                              const int* __restrict__ idx,
                              float* __restrict__ out, int rows) {
  int i = blockIdx.x * blockDim.x + threadIdx.x;
  if (i >= rows * 25) return;
  int r = i / 25, c = i % 25;
  ((float4*)out)[(size_t)r * 25 + c] =
      ((const float4*)emb)[(size_t)idx[r] * 25 + c];
}

// ---------------------------------------------------------------------------
// out[r][:] = mean_j emb[idx[r*S+j]][:]   (fused deepest-level gather+mean)
// ---------------------------------------------------------------------------
__global__ void gather_mean_kernel(const float* __restrict__ emb,
                                   const int* __restrict__ idx,
                                   float* __restrict__ out, int rows, int S) {
  int i = blockIdx.x * blockDim.x + threadIdx.x;
  if (i >= rows * 25) return;
  int r = i / 25, c = i % 25;
  float sx = 0.f, sy = 0.f, sz = 0.f, sw = 0.f;
  for (int j = 0; j < S; ++j) {
    float4 v = ((const float4*)emb)[(size_t)idx[r * S + j] * 25 + c];
    sx += v.x; sy += v.y; sz += v.z; sw += v.w;
  }
  float inv = 1.0f / (float)S;
  float4 o; o.x = sx * inv; o.y = sy * inv; o.z = sz * inv; o.w = sw * inv;
  ((float4*)out)[(size_t)r * 25 + c] = o;
}

// ---------------------------------------------------------------------------
// out[r][:] = mean_j in[r*S+j][:]
// ---------------------------------------------------------------------------
__global__ void mean_kernel(const float* __restrict__ in,
                            float* __restrict__ out, int rows, int S) {
  int i = blockIdx.x * blockDim.x + threadIdx.x;
  if (i >= rows * 25) return;
  int r = i / 25, c = i % 25;
  float sx = 0.f, sy = 0.f, sz = 0.f, sw = 0.f;
  for (int j = 0; j < S; ++j) {
    float4 v = ((const float4*)in)[((size_t)r * S + j) * 25 + c];
    sx += v.x; sy += v.y; sz += v.z; sw += v.w;
  }
  float inv = 1.0f / (float)S;
  float4 o; o.x = sx * inv; o.y = sy * inv; o.z = sz * inv; o.w = sw * inv;
  ((float4*)out)[(size_t)r * 25 + c] = o;
}

// ---------------------------------------------------------------------------
// Generic WMMA bf16 GEMM: Out[row, ocol+n] = act(A[row,:100] @ W + bias (+resid))
// LDS-free: one lane's A fragment for k-step s is exactly two contiguous
// 8-float runs of row (m): c in [32s+8hi, +8) and [32s+16+8hi, +8), loaded as
// four float4's and packed to bf16. B fragments are single 32B coalesced
// vector loads from the pre-swizzled weights. Branch-free templated epilogue.
// ---------------------------------------------------------------------------
template <bool HASBIAS, bool HASRESID, bool DORELU>
__global__ __launch_bounds__(32) void gemm_wmma_kernel(
    const float* __restrict__ A, int lda,
    const __bf16* __restrict__ Wsw, int ntiles,
    const float* __restrict__ bias,
    float* __restrict__ Out, int ldo, int ocol, int Nout,
    const float* __restrict__ resid, int ldr) {
  const int lane = threadIdx.x;
  const int m = lane & 15, hi = lane >> 4;
  const int row0 = blockIdx.x * 16;
  const float* rowp = A + (size_t)(row0 + m) * lda;

  v16bf afrag[4];
#pragma unroll
  for (int s = 0; s < 3; ++s) {            // k-steps 0..2: all c < 100
    float4 a0 = *(const float4*)(rowp + 32 * s + 8 * hi);
    float4 a1 = *(const float4*)(rowp + 32 * s + 8 * hi + 4);
    float4 b0 = *(const float4*)(rowp + 32 * s + 16 + 8 * hi);
    float4 b1 = *(const float4*)(rowp + 32 * s + 16 + 8 * hi + 4);
    pack8(afrag[s], 0, a0, a1);
    pack8(afrag[s], 8, b0, b1);
  }
  {                                        // k-step 3: only c = 96..99 valid
    float4 z; z.x = 0.f; z.y = 0.f; z.z = 0.f; z.w = 0.f;
    float4 a0 = (hi == 0) ? *(const float4*)(rowp + 96) : z;
    pack8(afrag[3], 0, a0, z);
    pack8(afrag[3], 8, z, z);
  }

  const v16bf* Wv = (const v16bf*)Wsw;
  for (int nt = 0; nt < ntiles; ++nt) {
    v8f acc = {0.f, 0.f, 0.f, 0.f, 0.f, 0.f, 0.f, 0.f};
#pragma unroll
    for (int s = 0; s < 4; ++s) {
      v16bf bfrag = Wv[(nt * 4 + s) * 32 + lane];       // 32B coalesced read
      acc = __builtin_amdgcn_wmma_f32_16x16x32_bf16(
          false, afrag[s], false, bfrag, (short)0, acc, false, false);
    }
    const int n = (nt << 4) + m;
    if (n < Nout) {
      float bv = HASBIAS ? bias[n] : 0.0f;
#pragma unroll
      for (int r = 0; r < 8; ++r) {
        int mm = r + 8 * hi;                            // documented C/D layout
        float v = acc[r] + bv;
        if (HASRESID) v += resid[(size_t)(row0 + mm) * ldr + ocol + n];
        if (DORELU) v = fmaxf(v, 0.0f);
        Out[(size_t)(row0 + mm) * ldo + ocol + n] = v;
      }
    }
  }
}

// ---------------------------------------------------------------------------
// Row-wise L2 normalize (eps 1e-12), one block per row.
// ---------------------------------------------------------------------------
__global__ void l2norm_kernel(const float* __restrict__ in,
                              float* __restrict__ out, int cols) {
  int r = blockIdx.x, t = threadIdx.x;
  __shared__ float red[128];
  float s = 0.0f;
  for (int c = t; c < cols; c += 128) { float x = in[(size_t)r * cols + c]; s += x * x; }
  red[t] = s; __syncthreads();
  for (int o = 64; o > 0; o >>= 1) { if (t < o) red[t] += red[t + o]; __syncthreads(); }
  float inv = 1.0f / fmaxf(sqrtf(red[0]), 1e-12f);
  for (int c = t; c < cols; c += 128)
    out[(size_t)r * cols + c] = in[(size_t)r * cols + c] * inv;
}

// ---------------------------------------------------------------------------
// LayerNorm over last dim (100), eps 1e-6, scale+shift. One block per row.
// ---------------------------------------------------------------------------
__global__ void layernorm_kernel(const float* __restrict__ in,
                                 float* __restrict__ out,
                                 const float* __restrict__ g,
                                 const float* __restrict__ b) {
  int r = blockIdx.x, t = threadIdx.x;
  __shared__ float red[128];
  float s = 0.0f;
  for (int c = t; c < DD; c += 128) s += in[(size_t)r * DD + c];
  red[t] = s; __syncthreads();
  for (int o = 64; o > 0; o >>= 1) { if (t < o) red[t] += red[t + o]; __syncthreads(); }
  float mu = red[0] * (1.0f / DD);
  __syncthreads();
  s = 0.0f;
  for (int c = t; c < DD; c += 128) { float x = in[(size_t)r * DD + c] - mu; s += x * x; }
  red[t] = s; __syncthreads();
  for (int o = 64; o > 0; o >>= 1) { if (t < o) red[t] += red[t + o]; __syncthreads(); }
  float inv = rsqrtf(red[0] * (1.0f / DD) + 1e-6f);
  for (int c = t; c < DD; c += 128)
    out[(size_t)r * DD + c] = (in[(size_t)r * DD + c] - mu) * inv * g[c] + b[c];
}

// ---------------------------------------------------------------------------
// 3x3 single-head attention per sample: O[b,g] = softmax(QK^T/10)[g,:] @ V
// ---------------------------------------------------------------------------
__global__ void attn_kernel(const float* __restrict__ Q,
                            const float* __restrict__ K,
                            const float* __restrict__ V,
                            float* __restrict__ O) {
  int b = blockIdx.x, t = threadIdx.x;   // 128 threads
  __shared__ float q[3][DD], k[3][DD], v[3][DD], a[3][3];
  for (int i = t; i < 3 * DD; i += 128) {
    int g = i / DD, d = i % DD;
    size_t off = ((size_t)b * 3 + g) * DD + d;
    q[g][d] = Q[off]; k[g][d] = K[off]; v[g][d] = V[off];
  }
  __syncthreads();
  if (t < 9) {
    int g = t / 3, h = t % 3;
    float s = 0.0f;
    for (int d = 0; d < DD; ++d) s += q[g][d] * k[h][d];
    a[g][h] = s * 0.1f;                 // 1/sqrt(100)
  }
  __syncthreads();
  if (t < 3) {
    float m = fmaxf(a[t][0], fmaxf(a[t][1], a[t][2]));
    float e0 = expf(a[t][0] - m), e1 = expf(a[t][1] - m), e2 = expf(a[t][2] - m);
    float inv = 1.0f / (e0 + e1 + e2);
    a[t][0] = e0 * inv; a[t][1] = e1 * inv; a[t][2] = e2 * inv;
  }
  __syncthreads();
  for (int i = t; i < 3 * DD; i += 128) {
    int g = i / DD, d = i % DD;
    O[((size_t)b * 3 + g) * DD + d] =
        a[g][0] * v[0][d] + a[g][1] * v[1][d] + a[g][2] * v[2][d];
  }
}

// ---------------------------------------------------------------------------
// out[b] = l2norm(base[nodeids[b]] + OE[edgetype[b]][b])   (width 200)
// ---------------------------------------------------------------------------
__global__ void final_kernel(const float* __restrict__ base,
                             const int* __restrict__ nodeids,
                             const int* __restrict__ et,
                             const float* __restrict__ OE,
                             float* __restrict__ out) {
  int b = blockIdx.x, t = threadIdx.x;   // 128 threads
  __shared__ float red[128];
  int nid = nodeids[b], e = et[b];
  int c0 = t, c1 = t + 128;
  float v0 = 0.0f, v1 = 0.0f, s = 0.0f;
  if (c0 < 200) { v0 = base[(size_t)nid * 200 + c0] + OE[((size_t)e * 512 + b) * 200 + c0]; s += v0 * v0; }
  if (c1 < 200) { v1 = base[(size_t)nid * 200 + c1] + OE[((size_t)e * 512 + b) * 200 + c1]; s += v1 * v1; }
  red[t] = s; __syncthreads();
  for (int o = 64; o > 0; o >>= 1) { if (t < o) red[t] += red[t + o]; __syncthreads(); }
  float inv = 1.0f / fmaxf(sqrtf(red[0]), 1e-12f);
  if (c0 < 200) out[(size_t)b * 200 + c0] = v0 * inv;
  if (c1 < 200) out[(size_t)b * 200 + c1] = v1 * inv;
}

// ---------------------------------------------------------------------------
extern "C" void kernel_launch(void* const* d_in, const int* in_sizes, int n_in,
                              void* d_out, int out_size, void* d_ws, size_t ws_size,
                              hipStream_t stream) {
  (void)in_sizes; (void)n_in; (void)out_size; (void)ws_size;
  const int*   nodeids = (const int*)d_in[0];
  const int*   edget   = (const int*)d_in[1];
  const int*   nl0     = (const int*)d_in[2];
  const int*   nl1     = (const int*)d_in[3];
  const int*   nl2     = (const int*)d_in[4];
  const int*   nl3     = (const int*)d_in[5];
  const float* base_w  = (const float*)d_in[6];
  const float* user_w  = (const float*)d_in[7];
  const float* W_self  = (const float*)d_in[8];
  const float* b_self  = (const float*)d_in[9];
  const float* W_neigh = (const float*)d_in[10];
  const float* b_neigh = (const float*)d_in[11];
  const float* W_fc    = (const float*)d_in[12];
  const float* b_fc    = (const float*)d_in[13];
  const float* ln_g    = (const float*)d_in[14];
  const float* ln_b    = (const float*)d_in[15];
  const float* W_q     = (const float*)d_in[16];
  const float* W_k     = (const float*)d_in[17];
  const float* W_v     = (const float*)d_in[18];
  const float* W_o     = (const float*)d_in[19];
  const float* reflect = (const float*)d_in[20];

  const int B = 512, G = 3, N = 100000;
  const int Ss[4]  = {3, 5, 7, 9};
  const int nsz[4] = {512, 1536, 7680, 53760};
  const int n3deep = 53760 * 9;  // 483840 (never materialized)

  // ---- ws bump allocator (256B aligned) ----
  char* ws = (char*)d_ws;
  size_t off = 0;
  auto alloc = [&](size_t bytes) -> void* {
    void* p = (void*)(ws + off);
    off += (bytes + 255) & ~(size_t)255;
    return p;
  };

  float* F[4]; float* M[4];
  for (int k = 0; k < 4; ++k) F[k] = (float*)alloc((size_t)nsz[k] * DD * 4);
  for (int k = 0; k < 4; ++k) M[k] = (float*)alloc((size_t)nsz[k] * DD * 4);
  float* NF0   = (float*)alloc((size_t)B * DD * 4);
  float* spec  = (float*)alloc((size_t)B * G * DD * 4);
  float* qn    = (float*)alloc((size_t)B * G * DD * 4);
  float* Qb    = (float*)alloc((size_t)B * G * DD * 4);
  float* Kb    = (float*)alloc((size_t)B * G * DD * 4);
  float* Vb    = (float*)alloc((size_t)B * G * DD * 4);
  float* Ob    = (float*)alloc((size_t)B * G * DD * 4);
  float* spec2 = (float*)alloc((size_t)B * G * DD * 4);
  float* OE    = (float*)alloc((size_t)G * B * 200 * 4);

  // swizzled bf16 weights: per matrix ntiles*2048 bf16
  __bf16* WsageP = (__bf16*)alloc((size_t)3 * 4 * 2 * 4 * 2048 * 2);   // ntiles=4
  __bf16* WfcP   = (__bf16*)alloc((size_t)3 * 8 * 2048 * 2);           // ntiles=8
  __bf16* WqP    = (__bf16*)alloc((size_t)8 * 2048 * 2);
  __bf16* WkP    = (__bf16*)alloc((size_t)8 * 2048 * 2);
  __bf16* WvP    = (__bf16*)alloc((size_t)8 * 2048 * 2);
  __bf16* WoP    = (__bf16*)alloc((size_t)8 * 2048 * 2);
  __bf16* ReflP  = (__bf16*)alloc((size_t)3 * 13 * 2048 * 2);          // ntiles=13

  auto pad = [&](const float* src, int Ka, int Na, __bf16* dst, int ntiles) {
    int tot = ntiles * 2048;
    padswz_kernel<<<(tot + 255) / 256, 256, 0, stream>>>(src, Ka, Na, dst, ntiles);
  };

  // ---- weight prep: pad K 100->128, convert bf16, pre-swizzle to frag order
  for (int g = 0; g < 3; ++g)
    for (int l = 0; l < 4; ++l) {
      size_t wi = ((size_t)(g * 4 + l) * 2);
      pad(W_self  + (size_t)(g * 4 + l) * DD * 50, DD, 50, WsageP + (wi + 0) * 4 * 2048, 4);
      pad(W_neigh + (size_t)(g * 4 + l) * DD * 50, DD, 50, WsageP + (wi + 1) * 4 * 2048, 4);
    }
  for (int g = 0; g < 3; ++g)
    pad(W_fc + (size_t)g * DD * DD, DD, DD, WfcP + (size_t)g * 8 * 2048, 8);
  pad(W_q, DD, DD, WqP, 8);
  pad(W_k, DD, DD, WkP, 8);
  pad(W_v, DD, DD, WvP, 8);
  pad(W_o, DD, DD, WoP, 8);
  for (int g = 0; g < 3; ++g)
    pad(reflect + (size_t)g * DD * 200, DD, 200, ReflP + (size_t)g * 13 * 2048, 13);

  // ---- GraphSage per graph type ----
  for (int g = 0; g < 3; ++g) {
    const float* emb = user_w + (size_t)g * N * DD;
    gather_kernel<<<(nsz[0] * 25 + 255) / 256, 256, 0, stream>>>(emb, nodeids, F[0], nsz[0]);
    gather_kernel<<<(nsz[1] * 25 + 255) / 256, 256, 0, stream>>>(emb, nl0 + (size_t)g * nsz[1], F[1], nsz[1]);
    gather_kernel<<<(nsz[2] * 25 + 255) / 256, 256, 0, stream>>>(emb, nl1 + (size_t)g * nsz[2], F[2], nsz[2]);
    gather_kernel<<<(nsz[3] * 25 + 255) / 256, 256, 0, stream>>>(emb, nl2 + (size_t)g * nsz[3], F[3], nsz[3]);

    for (int l = 0; l < 4; ++l) {
      int kmax = 4 - l;
      // means first (from current-layer inputs), deepest level fused with gather
      for (int k = 0; k < kmax; ++k) {
        int tot = nsz[k] * 25;
        if (l == 0 && k == 3) {
          gather_mean_kernel<<<(tot + 255) / 256, 256, 0, stream>>>(
              emb, nl3 + (size_t)g * n3deep, M[3], nsz[3], 9);
        } else {
          mean_kernel<<<(tot + 255) / 256, 256, 0, stream>>>(F[k + 1], M[k], nsz[k], Ss[k]);
        }
      }
      // dual-half GEMMs (in-place is safe: each wave reads its own 16 rows
      // into registers before any store to those rows)
      const __bf16* Wsp = WsageP + ((size_t)(g * 4 + l) * 2 + 0) * 4 * 2048;
      const __bf16* Wnp = WsageP + ((size_t)(g * 4 + l) * 2 + 1) * 4 * 2048;
      const float* bsp = b_self  + (size_t)(g * 4 + l) * 50;
      const float* bnp = b_neigh + (size_t)(g * 4 + l) * 50;
      for (int k = 0; k < kmax; ++k) {
        gemm_wmma_kernel<true, false, true><<<nsz[k] / 16, 32, 0, stream>>>(
            F[k], DD, Wsp, 4, bsp, F[k], DD, 0, 50, nullptr, 0);
        gemm_wmma_kernel<true, false, true><<<nsz[k] / 16, 32, 0, stream>>>(
            M[k], DD, Wnp, 4, bnp, F[k], DD, 50, 50, nullptr, 0);
      }
    }
    l2norm_kernel<<<B, 128, 0, stream>>>(F[0], NF0, DD);
    gemm_wmma_kernel<true, false, false><<<B / 16, 32, 0, stream>>>(
        NF0, DD, WfcP + (size_t)g * 8 * 2048, 8, b_fc + (size_t)g * DD,
        spec, G * DD, g * DD, DD, nullptr, 0);
  }

  // ---- attention across graph types ----
  layernorm_kernel<<<B * G, 128, 0, stream>>>(spec, qn, ln_g, ln_b);
  gemm_wmma_kernel<false, false, false><<<(B * G) / 16, 32, 0, stream>>>(
      qn, DD, WqP, 8, nullptr, Qb, DD, 0, DD, nullptr, 0);
  gemm_wmma_kernel<false, false, false><<<(B * G) / 16, 32, 0, stream>>>(
      spec, DD, WkP, 8, nullptr, Kb, DD, 0, DD, nullptr, 0);
  gemm_wmma_kernel<false, false, false><<<(B * G) / 16, 32, 0, stream>>>(
      spec, DD, WvP, 8, nullptr, Vb, DD, 0, DD, nullptr, 0);
  attn_kernel<<<B, 128, 0, stream>>>(Qb, Kb, Vb, Ob);
  gemm_wmma_kernel<false, true, false><<<(B * G) / 16, 32, 0, stream>>>(
      Ob, DD, WoP, 8, nullptr, spec2, DD, 0, DD, spec, DD);

  // ---- reflect matmul for all 3 edge types, then gather-select ----
  for (int g = 0; g < 3; ++g)
    gemm_wmma_kernel<false, false, false><<<B / 16, 32, 0, stream>>>(
        spec2 + (size_t)g * DD, G * DD, ReflP + (size_t)g * 13 * 2048, 13,
        nullptr, OE + (size_t)g * B * 200, 200, 0, 200, nullptr, 0);

  final_kernel<<<B, 128, 0, stream>>>(base_w, nodeids, edget, OE, (float*)d_out);
}
